// Model_49675591746044
// MI455X (gfx1250) — compile-verified, matrix operand order checked by reading, not run on hardware
//
#include <hip/hip_runtime.h>
#include <hip/hip_bf16.h>

// ---------------------------------------------------------------------------
// VQ-VAE decoder forward for MI455X (gfx1250, wave32, WMMA + TDM).
// GEMMs: 128x128 workgroup tiles, TDM (tensor_load_to_lds) stages fp32 tiles
// into LDS with hardware padding (33-float row stride), fragments converted
// fp32->bf16 on the fly, math on v_wmma_f32_16x16x32_bf16 (f32 accumulate).
// Softmax / LayerNorm / top-k are fp32 VALU.
// ---------------------------------------------------------------------------

typedef __attribute__((ext_vector_type(16))) __bf16 v16bf;
typedef __attribute__((ext_vector_type(8)))  float  v8f;
typedef __attribute__((ext_vector_type(4)))  unsigned v4u;
typedef __attribute__((ext_vector_type(4)))  int    v4i;
typedef __attribute__((ext_vector_type(8)))  int    v8i;

#define BN   16
#define SN   512
#define DN   768
#define KN   8192
#define HN   8
#define LN   6
#define FFN  2048
#define NTOK (BN * SN)      // 8192
#define D3   (3 * DN)       // 2304
#define DH   96             // head dim
#define NEGB (-1e9f)

#if defined(__HIP_DEVICE_COMPILE__) && __has_builtin(__builtin_amdgcn_tensor_load_to_lds)
#define USE_TDM 1
#else
#define USE_TDM 0
#endif

// fp32 -> bf16, round-to-nearest-even
__device__ __forceinline__ unsigned short f2bfu(float f) {
  unsigned u = __builtin_bit_cast(unsigned, f);
  u = (u + 0x7FFFu + ((u >> 16) & 1u)) >> 16;
  return (unsigned short)u;
}
__device__ __forceinline__ __bf16 us2bf(unsigned short v) {
  return __builtin_bit_cast(__bf16, v);
}

// --- WMMA fragment gathers (ISA 7.12.2 layouts) ----------------------------
// A 16x32 bf16 from LDS stored row-major [16 rows][>=32 k] (bf16 source).
__device__ __forceinline__ v16bf frag_nk(const unsigned short* p, int ld) {
  int lane = threadIdx.x & 31;
  const unsigned short* r = p + (size_t)(lane & 15) * ld;
  int khi = (lane >> 4) << 3;  // 0 or 8
  v16bf a;
#pragma unroll
  for (int v = 0; v < 8; ++v) {
    int kb = ((v & 4) << 2) + khi + ((v & 3) << 1);
    a[2 * v]     = us2bf(r[kb]);
    a[2 * v + 1] = us2bf(r[kb + 1]);
  }
  return a;
}
// Same layout but fp32 LDS source, converted to bf16 while gathering.
__device__ __forceinline__ v16bf frag_nk_f32(const float* p, int ld) {
  int lane = threadIdx.x & 31;
  const float* r = p + (size_t)(lane & 15) * ld;
  int khi = (lane >> 4) << 3;
  v16bf a;
#pragma unroll
  for (int v = 0; v < 8; ++v) {
    int kb = ((v & 4) << 2) + khi + ((v & 3) << 1);
    a[2 * v]     = us2bf(f2bfu(r[kb]));
    a[2 * v + 1] = us2bf(f2bfu(r[kb + 1]));
  }
  return a;
}
// B 32x16 (KxN) where LDS tile is stored [32 k-rows][>=16 n] (bf16 source).
__device__ __forceinline__ v16bf frag_kn(const unsigned short* p, int ld) {
  int lane = threadIdx.x & 31;
  int n = lane & 15;
  int khi = (lane >> 4) << 3;
  v16bf a;
#pragma unroll
  for (int v = 0; v < 8; ++v) {
    int kb = ((v & 4) << 2) + khi + ((v & 3) << 1);
    a[2 * v]     = us2bf(p[(size_t)kb * ld + n]);
    a[2 * v + 1] = us2bf(p[(size_t)(kb + 1) * ld + n]);
  }
  return a;
}

__device__ __forceinline__ v8f zero8() {
  v8f z = {0.f, 0.f, 0.f, 0.f, 0.f, 0.f, 0.f, 0.f};
  return z;
}

// hash -> uniform (0,1)  (reference uses jax threefry; unreproducible bitwise)
__device__ __forceinline__ float u01(unsigned x) {
  x ^= x >> 17; x *= 0xed5ad4bbu; x ^= x >> 11; x *= 0xac4c1b51u;
  x ^= x >> 15; x *= 0x31848babu; x ^= x >> 14;
  return ((float)(x >> 8) + 0.5f) * (1.0f / 16777216.0f);
}

#if USE_TDM
// Issue one TDM 2-D tile load: rows x 32 fp32, global row stride = ld elts,
// LDS padded 1 DWORD per 32 DWORDs (=> 33-float LDS row stride).
__device__ __forceinline__ void tdm_load_tile_f32(unsigned lds_off,
                                                  const float* gsrc,
                                                  int ld, int rows) {
  unsigned long long ga = (unsigned long long)(size_t)gsrc;
  v4u g0;
  g0[0] = 1u;                                    // count=1, user mode
  g0[1] = lds_off;                               // lds_addr (bytes)
  g0[2] = (unsigned)(ga & 0xffffffffu);          // global_addr[31:0]
  g0[3] = (unsigned)((ga >> 32) & 0x1ffffffu)    // global_addr[56:32]
          | (2u << 30);                          // type=2 ("image")
  v8i g1;
  g1[0] = (int)((2u << 16)      // data_size = 4B
              | (1u << 20)      // pad_enable
              | (4u << 22));    // pad_interval: 32 DWORDs
  g1[1] = 0;                                       // tensor_dim0 lo16 in [31:16]
  g1[2] = 0x4000;                                  // tensor_dim0 = 2^30 (no OOB)
  g1[3] = (int)(0x4000u | (32u << 16));            // tensor_dim1 = 2^30; tile_dim0=32
  g1[4] = rows;                                    // tile_dim1 = rows; tile_dim2=0
  g1[5] = ld;                                      // tensor_dim0_stride lo32
  g1[6] = 0;
  g1[7] = 0;
  v4i z4 = {0, 0, 0, 0};
#if __clang_major__ >= 23
  v8i z8 = {0, 0, 0, 0, 0, 0, 0, 0};
  __builtin_amdgcn_tensor_load_to_lds(g0, g1, z4, z4, z8, 0);
#else
  __builtin_amdgcn_tensor_load_to_lds(g0, g1, z4, z4, 0);
#endif
}
#endif

// ---------------------------------------------------------------------------
// 1) padding mask + z_e
__global__ void pad_ze_kernel(const float* __restrict__ x,
                              float* __restrict__ ze,
                              float* __restrict__ padneg) {
  __shared__ float red[256];
  int t = blockIdx.x, tid = threadIdx.x;
  float v[3]; float ss = 0.f;
#pragma unroll
  for (int j = 0; j < 3; ++j) {
    v[j] = x[(size_t)t * DN + tid + 256 * j];
    ss += v[j] * v[j];
  }
  red[tid] = ss; __syncthreads();
  for (int s = 128; s > 0; s >>= 1) { if (tid < s) red[tid] += red[tid + s]; __syncthreads(); }
  float tot = red[0];
  bool pad = (tot <= 1e-12f);   // norm <= 1e-6
  if (tid == 0) padneg[t] = pad ? NEGB : 0.f;
#pragma unroll
  for (int j = 0; j < 3; ++j)
    ze[(size_t)t * DN + tid + 256 * j] = pad ? 0.f : v[j];
}

// 2) codebook row norms^2
__global__ void cnorm_kernel(const float* __restrict__ cb, float* __restrict__ cn2) {
  int c = blockIdx.x * blockDim.x + threadIdx.x;
  if (c >= KN) return;
  float s = 0.f;
  for (int i = 0; i < DN; ++i) { float v = cb[(size_t)c * DN + i]; s += v * v; }
  cn2[c] = s;
}

// 3) fused VQ: 16-token tile per wave, top-10 + gumbel-argmax -> enc
__global__ void vq_topk_kernel(const float* __restrict__ ze,
                               const float* __restrict__ cb,
                               const float* __restrict__ cn2,
                               int* __restrict__ enc) {
  __shared__ unsigned short Xs[16 * 776];   // 16 x 768 bf16 (padded)
  __shared__ unsigned short Cs[16 * 34];    // 16 codes x 32 k
  __shared__ float Ds[16 * 16];
  int lane = threadIdx.x & 31;
  int t0 = blockIdx.x * 16;
  for (int i = lane; i < 16 * DN; i += 32) {
    int r = i / DN, k = i - r * DN;
    Xs[r * 776 + k] = f2bfu(ze[(size_t)(t0 + r) * DN + k]);
  }
  __syncthreads();

  float best[10]; int bidx[10];
#pragma unroll
  for (int j = 0; j < 10; ++j) { best[j] = 3e38f; bidx[j] = 0; }

  for (int c0 = 0; c0 < KN; c0 += 16) {
    v8f acc = zero8();
    for (int kc = 0; kc < DN; kc += 32) {
      __syncthreads();
      for (int i = lane; i < 16 * 32; i += 32) {
        int r = i >> 5, k = i & 31;
        Cs[r * 34 + k] = f2bfu(cb[(size_t)(c0 + r) * DN + kc + k]);
      }
      __syncthreads();
      v16bf a = frag_nk(Xs + kc, 776);
      v16bf b = frag_nk(Cs, 34);
      acc = __builtin_amdgcn_wmma_f32_16x16x32_bf16(false, a, false, b,
                                                    (short)0, acc, false, false);
    }
#pragma unroll
    for (int v = 0; v < 8; ++v) {
      int tr = v + ((lane >> 4) << 3);
      int cc = lane & 15;
      Ds[tr * 16 + cc] = cn2[c0 + cc] - 2.f * acc[v];  // ||x||^2 dropped (invariant)
    }
    __syncthreads();
    if (lane < 16) {
      for (int j = 0; j < 16; ++j) {
        float d = Ds[lane * 16 + j];
        if (d < best[9]) {
          int p = 9;
          while (p > 0 && best[p - 1] > d) { best[p] = best[p - 1]; bidx[p] = bidx[p - 1]; --p; }
          best[p] = d; bidx[p] = c0 + j;
        }
      }
    }
    __syncthreads();
  }
  if (lane < 16) {
    int token = t0 + lane;
    float bs = -3e38f; int bi = bidx[0];
#pragma unroll
    for (int j = 0; j < 10; ++j) {
      float u = u01((unsigned)(token * 10 + j) * 2654435761u + 42u);
      float g = -__logf(-__logf(u));
      float sc = -best[j] + g;      // TEMP = 1
      if (sc > bs) { bs = sc; bi = bidx[j]; }
    }
    enc[token] = bi;
  }
}

// 4) quantized = codebook[enc]
__global__ void gather_kernel(const float* __restrict__ cb,
                              const int* __restrict__ enc,
                              float* __restrict__ h) {
  size_t idx = (size_t)blockIdx.x * blockDim.x + threadIdx.x;
  int t = (int)(idx / DN), k = (int)(idx - (size_t)t * DN);
  h[idx] = cb[(size_t)enc[t] * DN + k];
}

// 5) WMMA GEMM: C[M,N] = act(A[M,kd] @ W[N,kd]^T + bias)
//    256 threads (8 waves); WG tile 128x128; wave -> 32 rows x 64 cols.
//    TDM stages both fp32 tiles into padded LDS (stride 33 floats).
__global__ void gemm128_kernel(const float* __restrict__ A, int lda,
                               const float* __restrict__ W, int ldw,
                               const float* __restrict__ bias,
                               float* __restrict__ C, int ldc,
                               int kd, int relu) {
  __shared__ float As[128 * 33];
  __shared__ float Bs[128 * 33];
  int tid = threadIdx.x, wv = tid >> 5, lane = tid & 31;
  int mw = wv >> 1, nw = wv & 1;
  int m0 = blockIdx.y * 128, n0 = blockIdx.x * 128;
  v8f acc[2][4];
#pragma unroll
  for (int mi = 0; mi < 2; ++mi)
#pragma unroll
    for (int nj = 0; nj < 4; ++nj) acc[mi][nj] = zero8();

#if USE_TDM
  unsigned As_off = (unsigned)(size_t)&As[0];   // LDS aperture: low 32 bits = LDS offset
  unsigned Bs_off = (unsigned)(size_t)&Bs[0];
#endif

  for (int k0 = 0; k0 < kd; k0 += 32) {
    __syncthreads();
#if USE_TDM
    if (wv == 0) {
      tdm_load_tile_f32(As_off, &A[(size_t)m0 * lda + k0], lda, 128);
      tdm_load_tile_f32(Bs_off, &W[(size_t)n0 * ldw + k0], ldw, 128);
      __builtin_amdgcn_s_wait_tensorcnt(0);
    }
#else
    for (int i = tid; i < 128 * 32; i += 256) {
      int r = i >> 5, k = i & 31;
      As[r * 33 + k] = A[(size_t)(m0 + r) * lda + k0 + k];
      Bs[r * 33 + k] = W[(size_t)(n0 + r) * ldw + k0 + k];
    }
    if (k0 + 32 < kd) {
      __builtin_prefetch(&A[(size_t)(m0 + (tid >> 1)) * lda + k0 + 32], 0, 1);
      __builtin_prefetch(&W[(size_t)(n0 + (tid >> 1)) * ldw + k0 + 32], 0, 1);
    }
#endif
    __syncthreads();
    v16bf af[2];
#pragma unroll
    for (int mi = 0; mi < 2; ++mi)
      af[mi] = frag_nk_f32(As + (size_t)(mw * 32 + mi * 16) * 33, 33);
#pragma unroll
    for (int nj = 0; nj < 4; ++nj) {
      v16bf bfr = frag_nk_f32(Bs + (size_t)(nw * 64 + nj * 16) * 33, 33);
#pragma unroll
      for (int mi = 0; mi < 2; ++mi)
        acc[mi][nj] = __builtin_amdgcn_wmma_f32_16x16x32_bf16(
            false, af[mi], false, bfr, (short)0, acc[mi][nj], false, false);
    }
  }
#pragma unroll
  for (int mi = 0; mi < 2; ++mi)
#pragma unroll
    for (int nj = 0; nj < 4; ++nj)
#pragma unroll
      for (int v = 0; v < 8; ++v) {
        int row = m0 + mw * 32 + mi * 16 + v + ((lane >> 4) << 3);
        int col = n0 + nw * 64 + nj * 16 + (lane & 15);
        float x = acc[mi][nj][v] + bias[col];
        if (relu) x = fmaxf(x, 0.f);
        C[(size_t)row * ldc + col] = x;
      }
}

// 6) flash attention (unchanged from round 1; bf16 WMMA, fp32 online softmax)
__global__ void attn_kernel(const float* __restrict__ qkv,
                            const float* __restrict__ padneg,
                            float* __restrict__ out, int causal) {
  __shared__ unsigned short Qs[16 * 104];
  __shared__ unsigned short KVs[32 * 104];
  __shared__ unsigned short Ps[16 * 40];
  int lane = threadIdx.x & 31;
  int b = blockIdx.z, hd = blockIdx.y, q0 = blockIdx.x * 16;
  int qcol = hd * DH, kcol = DN + hd * DH, vcol = 2 * DN + hd * DH;
  const float scale = 0.10206207262f;  // 1/sqrt(96)

  for (int i = lane; i < 16 * DH; i += 32) {
    int r = i / DH, c = i - r * DH;
    Qs[r * 104 + c] = f2bfu(qkv[(size_t)(b * SN + q0 + r) * D3 + qcol + c]);
  }
  __syncthreads();
  v16bf aq[3];
#pragma unroll
  for (int c = 0; c < 3; ++c) aq[c] = frag_nk(Qs + c * 32, 104);

  v8f o[6];
#pragma unroll
  for (int t = 0; t < 6; ++t) o[t] = zero8();
  float m8[8], l8[8];
#pragma unroll
  for (int v = 0; v < 8; ++v) { m8[v] = -3e38f; l8[v] = 0.f; }

  for (int kc = 0; kc < SN; kc += 32) {
    __syncthreads();
    for (int i = lane; i < 32 * DH; i += 32) {
      int r = i / DH, c = i - r * DH;
      KVs[r * 104 + c] = f2bfu(qkv[(size_t)(b * SN + kc + r) * D3 + kcol + c]);
    }
    __syncthreads();
    v8f s0 = zero8(), s1 = zero8();
#pragma unroll
    for (int c = 0; c < 3; ++c) {
      v16bf bk0 = frag_nk(KVs + c * 32, 104);
      v16bf bk1 = frag_nk(KVs + 16 * 104 + c * 32, 104);
      s0 = __builtin_amdgcn_wmma_f32_16x16x32_bf16(false, aq[c], false, bk0,
                                                   (short)0, s0, false, false);
      s1 = __builtin_amdgcn_wmma_f32_16x16x32_bf16(false, aq[c], false, bk1,
                                                   (short)0, s1, false, false);
    }
#pragma unroll
    for (int v = 0; v < 8; ++v) {
      int q = q0 + v + ((lane >> 4) << 3);
      int k0i = kc + (lane & 15), k1i = k0i + 16;
      float x0 = s0[v] * scale + padneg[b * SN + k0i] +
                 ((causal && k0i < q) ? NEGB : 0.f);
      float x1 = s1[v] * scale + padneg[b * SN + k1i] +
                 ((causal && k1i < q) ? NEGB : 0.f);
      float mx = fmaxf(x0, x1);
#pragma unroll
      for (int dd = 8; dd; dd >>= 1) mx = fmaxf(mx, __shfl_xor(mx, dd, 16));
      float mnew = fmaxf(m8[v], mx);
      float alpha = __expf(m8[v] - mnew);
      m8[v] = mnew;
      float p0 = __expf(x0 - mnew), p1 = __expf(x1 - mnew);
      s0[v] = p0; s1[v] = p1;
      float rs = p0 + p1;
#pragma unroll
      for (int dd = 8; dd; dd >>= 1) rs += __shfl_xor(rs, dd, 16);
      l8[v] = l8[v] * alpha + rs;
#pragma unroll
      for (int t = 0; t < 6; ++t) o[t][v] *= alpha;
    }
    __syncthreads();
#pragma unroll
    for (int v = 0; v < 8; ++v) {
      int tr = v + ((lane >> 4) << 3);
      Ps[tr * 40 + (lane & 15)] = f2bfu(s0[v]);
      Ps[tr * 40 + 16 + (lane & 15)] = f2bfu(s1[v]);
    }
    for (int i = lane; i < 32 * DH; i += 32) {
      int r = i / DH, c = i - r * DH;
      KVs[r * 104 + c] = f2bfu(qkv[(size_t)(b * SN + kc + r) * D3 + vcol + c]);
    }
    __syncthreads();
    v16bf ap = frag_nk(Ps, 40);
#pragma unroll
    for (int t = 0; t < 6; ++t) {
      v16bf bv = frag_kn(KVs + t * 16, 104);
      o[t] = __builtin_amdgcn_wmma_f32_16x16x32_bf16(false, ap, false, bv,
                                                     (short)0, o[t], false, false);
    }
  }
#pragma unroll
  for (int t = 0; t < 6; ++t)
#pragma unroll
    for (int v = 0; v < 8; ++v) {
      int q = q0 + v + ((lane >> 4) << 3);
      float denom = (l8[v] > 0.f) ? l8[v] : 1.f;
      out[(size_t)(b * SN + q) * DN + hd * DH + t * 16 + (lane & 15)] = o[t][v] / denom;
    }
}

// 7) y = LayerNorm(x + r) * g + b
__global__ void add_ln_kernel(const float* __restrict__ x,
                              const float* __restrict__ r,
                              const float* __restrict__ g,
                              const float* __restrict__ bt,
                              float* __restrict__ y) {
  __shared__ float red[256];
  int t = blockIdx.x, tid = threadIdx.x;
  float v[3]; float s = 0.f;
#pragma unroll
  for (int j = 0; j < 3; ++j) {
    v[j] = x[(size_t)t * DN + tid + 256 * j] + r[(size_t)t * DN + tid + 256 * j];
    s += v[j];
  }
  red[tid] = s; __syncthreads();
  for (int st = 128; st > 0; st >>= 1) { if (tid < st) red[tid] += red[tid + st]; __syncthreads(); }
  float mean = red[0] * (1.f / DN); __syncthreads();
  float vs = 0.f;
#pragma unroll
  for (int j = 0; j < 3; ++j) { float d = v[j] - mean; vs += d * d; }
  red[tid] = vs; __syncthreads();
  for (int st = 128; st > 0; st >>= 1) { if (tid < st) red[tid] += red[tid + st]; __syncthreads(); }
  float inv = __frsqrt_rn(red[0] * (1.f / DN) + 1e-5f);
#pragma unroll
  for (int j = 0; j < 3; ++j) {
    int c = tid + 256 * j;
    y[(size_t)t * DN + c] = (v[j] - mean) * inv * g[c] + bt[c];
  }
}

// ---------------------------------------------------------------------------
extern "C" void kernel_launch(void* const* d_in, const int* in_sizes, int n_in,
                              void* d_out, int out_size, void* d_ws, size_t ws_size,
                              hipStream_t stream) {
  const float* x        = (const float*)d_in[0];
  const float* cb       = (const float*)d_in[1];
  const float* sa_in_w  = (const float*)d_in[2];
  const float* sa_in_b  = (const float*)d_in[3];
  const float* sa_out_w = (const float*)d_in[4];
  const float* sa_out_b = (const float*)d_in[5];
  const float* ca_in_w  = (const float*)d_in[6];
  const float* ca_in_b  = (const float*)d_in[7];
  const float* ca_out_w = (const float*)d_in[8];
  const float* ca_out_b = (const float*)d_in[9];
  const float* ln1_g = (const float*)d_in[10];
  const float* ln1_b = (const float*)d_in[11];
  const float* ln2_g = (const float*)d_in[12];
  const float* ln2_b = (const float*)d_in[13];
  const float* ln3_g = (const float*)d_in[14];
  const float* ln3_b = (const float*)d_in[15];
  const float* ff_w1 = (const float*)d_in[16];
  const float* ff_b1 = (const float*)d_in[17];
  const float* ff_w2 = (const float*)d_in[18];
  const float* ff_b2 = (const float*)d_in[19];
  const float* out_w = (const float*)d_in[20];
  const float* out_b = (const float*)d_in[21];

  char* ws = (char*)d_ws;
  size_t o = 0;
  auto alloc = [&](size_t bytes) { char* p = ws + o; o += (bytes + 255) & ~(size_t)255; return p; };
  float* ze     = (float*)alloc((size_t)NTOK * DN * 4);
  float* h      = (float*)alloc((size_t)NTOK * DN * 4);
  float* qkv    = (float*)alloc((size_t)NTOK * D3 * 4);   // reused for FF hidden
  float* attn_o = (float*)alloc((size_t)NTOK * DN * 4);
  float* tmp    = (float*)alloc((size_t)NTOK * DN * 4);
  float* padneg = (float*)alloc((size_t)NTOK * 4);
  float* cn2    = (float*)alloc((size_t)KN * 4);
  int*   enc    = (int*)alloc((size_t)NTOK * 4);
  float* ffbuf  = qkv;

  // VQ front-end
  pad_ze_kernel<<<NTOK, 256, 0, stream>>>(x, ze, padneg);
  cnorm_kernel<<<KN / 256, 256, 0, stream>>>(cb, cn2);
  vq_topk_kernel<<<NTOK / 16, 32, 0, stream>>>(ze, cb, cn2, enc);
  gather_kernel<<<(NTOK * DN) / 256, 256, 0, stream>>>(cb, enc, h);

  const int GM = NTOK / 128;   // 64
  for (int i = 0; i < LN; ++i) {
    const float* saw = sa_in_w + (size_t)i * D3 * DN;
    const float* sab = sa_in_b + (size_t)i * D3;
    const float* sow = sa_out_w + (size_t)i * DN * DN;
    const float* sob = sa_out_b + (size_t)i * DN;
    const float* caw = ca_in_w + (size_t)i * D3 * DN;
    const float* cabv = ca_in_b + (size_t)i * D3;
    const float* cow = ca_out_w + (size_t)i * DN * DN;
    const float* cob = ca_out_b + (size_t)i * DN;

    // ---- self attention ----
    gemm128_kernel<<<dim3(D3 / 128, GM), 256, 0, stream>>>(
        h, DN, saw, DN, sab, qkv, D3, DN, 0);
    attn_kernel<<<dim3(SN / 16, HN, BN), 32, 0, stream>>>(qkv, padneg, attn_o, 1);
    gemm128_kernel<<<dim3(DN / 128, GM), 256, 0, stream>>>(
        attn_o, DN, sow, DN, sob, tmp, DN, DN, 0);
    add_ln_kernel<<<NTOK, 256, 0, stream>>>(h, tmp,
        ln1_g + (size_t)i * DN, ln1_b + (size_t)i * DN, h);

    // ---- cross attention: Q from h, K/V from mem (z_e) ----
    gemm128_kernel<<<dim3(DN / 128, GM), 256, 0, stream>>>(
        h, DN, caw, DN, cabv, qkv, D3, DN, 0);
    gemm128_kernel<<<dim3((2 * DN) / 128, GM), 256, 0, stream>>>(
        ze, DN, caw + (size_t)DN * DN, DN, cabv + DN, qkv + DN, D3, DN, 0);
    attn_kernel<<<dim3(SN / 16, HN, BN), 32, 0, stream>>>(qkv, padneg, attn_o, 0);
    gemm128_kernel<<<dim3(DN / 128, GM), 256, 0, stream>>>(
        attn_o, DN, cow, DN, cob, tmp, DN, DN, 0);
    add_ln_kernel<<<NTOK, 256, 0, stream>>>(h, tmp,
        ln2_g + (size_t)i * DN, ln2_b + (size_t)i * DN, h);

    // ---- feed forward ----
    gemm128_kernel<<<dim3(FFN / 128, GM), 256, 0, stream>>>(
        h, DN, ff_w1 + (size_t)i * FFN * DN, DN, ff_b1 + (size_t)i * FFN,
        ffbuf, FFN, DN, 1);
    gemm128_kernel<<<dim3(DN / 128, GM), 256, 0, stream>>>(
        ffbuf, FFN, ff_w2 + (size_t)i * DN * FFN, FFN, ff_b2 + (size_t)i * DN,
        tmp, DN, FFN, 0);
    add_ln_kernel<<<NTOK, 256, 0, stream>>>(h, tmp,
        ln3_g + (size_t)i * DN, ln3_b + (size_t)i * DN, h);
  }

  gemm128_kernel<<<dim3(DN / 128, GM), 256, 0, stream>>>(
      h, DN, out_w, DN, out_b, (float*)d_out, DN, DN, 0);
  (void)in_sizes; (void)n_in; (void)out_size; (void)ws_size;
}
